// CrossBidirectionalAttention_22986664969098
// MI455X (gfx1250) — matrix operand order repaired and checked by reading, not compiled, not run on hardware
//
#include <hip/hip_runtime.h>
#include <hip/hip_bf16.h>
#include <math.h>

// ---------------------------------------------------------------------------
// CrossBidirectionalAttention for MI455X (gfx1250, wave32, WMMA f16 16x16x32)
// B=2, L=S=2048, C=1024, H=16, HD=64
//   1) f32 -> f16 casts
//   2) WMMA NT-GEMMs (async-LDS double-buffered weight tiles, ping-pong regs)
//   3) flash attention x2 (async-LDS double-buffered K tiles, online softmax)
//   4) WMMA NT-GEMMs: o @ Wmerge^T -> f32 d_out
// ---------------------------------------------------------------------------

typedef _Float16 half_t;
typedef __attribute__((ext_vector_type(16))) _Float16 v16h;
typedef __attribute__((ext_vector_type(8)))  _Float16 v8h;
typedef __attribute__((ext_vector_type(8)))  float    v8f;

#define WMMA_F32F16(a, b, c) \
  __builtin_amdgcn_wmma_f32_16x16x32_f16(false, (a), false, (b), (short)0, (c), false, false)

// A fragment (16x32 f16, M x K): lane m = lane&15; hi = lane>>4
//   elems 0..7  -> K =  8*hi + i      (contiguous 8 halfs)
//   elems 8..15 -> K = 16 + 8*hi + i  (contiguous 8 halfs)
static __device__ __forceinline__ v16h load_frag_a(const half_t* __restrict__ src, int ld) {
  const int lane = threadIdx.x & 31;
  const int m = lane & 15, hi = lane >> 4;
  const half_t* r = src + (size_t)m * ld;
  v8h lo  = *(const v8h*)(r + 8 * hi);
  v8h hi8 = *(const v8h*)(r + 16 + 8 * hi);
  v16h out;
#pragma unroll
  for (int i = 0; i < 8; ++i) { out[i] = lo[i]; out[i + 8] = hi8[i]; }
  return out;
}

// B fragment (32x16 f16, K x N), source rows indexed by N with K on fast axis:
//   lane n = lane&15; elems 0..7 -> K = 8*hi + i, elems 8..15 -> K = 16 + 8*hi + i
static __device__ __forceinline__ v16h load_frag_b(const half_t* __restrict__ src, int ld) {
  const int lane = threadIdx.x & 31;
  const int n = lane & 15, hi = lane >> 4;
  const half_t* r = src + (size_t)n * ld;
  v8h lo  = *(const v8h*)(r + 8 * hi);
  v8h hi8 = *(const v8h*)(r + 16 + 8 * hi);
  v16h out;
#pragma unroll
  for (int i = 0; i < 8; ++i) { out[i] = lo[i]; out[i + 8] = hi8[i]; }
  return out;
}

// CDNA5 async copy: 16 bytes global -> LDS, tracked by ASYNCcnt.
static __device__ __forceinline__ void async_copy_b128(const half_t* g, half_t* l) {
  const unsigned lds_off = (unsigned)(size_t)l;  // low 32 bits of generic shared ptr = LDS offset
  asm volatile("global_load_async_to_lds_b128 %0, %1, off" :: "v"(lds_off), "v"(g) : "memory");
}
static __device__ __forceinline__ void wait_async() {
  asm volatile("s_wait_asynccnt 0x0" ::: "memory");
}

// ---------------------------------------------------------------------------
__global__ void cast_f32_to_f16(const float* __restrict__ src, half_t* __restrict__ dst, int n) {
  int i = (blockIdx.x * blockDim.x + threadIdx.x) * 8;
  if (i + 8 <= n) {
#pragma unroll
    for (int k = 0; k < 8; ++k) dst[i + k] = (half_t)src[i + k];
  }
}

// ---------------------------------------------------------------------------
// out[M,N] = scale * (A[M,K] @ Bw[N,K]^T);  A,Bw f16 row-major.  K % 64 == 0.
// Block: 128 threads (4 waves); tile 128x64 (each wave 32 rows x 64 cols).
// Weight tiles (64 N-rows x 32 K) double-buffered in LDS via async b128 copies;
// A fragments ping-pong between even/odd register sets (manual 2x unroll).
__global__ void gemm_nt_wmma(const half_t* __restrict__ A, const half_t* __restrict__ Bw,
                             void* __restrict__ out, int M, int N, int K,
                             float scale, int out_is_f32) {
  constexpr int LDB = 40;  // halfs per LDS B row: 80B, 16B-aligned, bank-skewed
  __shared__ __align__(16) half_t Bs[2][64 * LDB];

  const int wave = threadIdx.x >> 5;
  const int lane = threadIdx.x & 31;
  const int hi = lane >> 4, nl = lane & 15;
  const int n0 = blockIdx.x * 64;
  const int m0 = blockIdx.y * 128 + wave * 32;

  // this thread's two 16B chunks of the 64x32 B tile (64B per row, 256 chunks)
  const int c0 = threadIdx.x,       row0 = c0 >> 2, seg0 = (c0 & 3) * 8;
  const int c1 = threadIdx.x + 128, row1 = c1 >> 2, seg1 = (c1 & 3) * 8;
  const half_t* gB0 = Bw + (size_t)(n0 + row0) * K + seg0;
  const half_t* gB1 = Bw + (size_t)(n0 + row1) * K + seg1;
  half_t* lB0 = &Bs[0][row0 * LDB + seg0];
  half_t* lB1 = &Bs[0][row1 * LDB + seg1];
  const size_t lbuf = 64 * LDB;  // halfs between the two LDS buffers

  // prologue: stage k=0 into buffer 0
  async_copy_b128(gB0, lB0);
  async_copy_b128(gB1, lB1);
  wait_async();
  __syncthreads();

  v8f acc[2][4] = {};
  const half_t* Arow0 = A + (size_t)m0 * K;
  const half_t* Arow1 = A + (size_t)(m0 + 16) * K;
  v16h aE0 = load_frag_a(Arow0, K);
  v16h aE1 = load_frag_a(Arow1, K);
  v16h aO0, aO1;

  for (int k0 = 0; k0 < K; k0 += 64) {
    // ---- phase 0: compute Bs[0] with aE*, stage k0+32 -> Bs[1] ----
    {
      const int kn = k0 + 32;  // kn < K since K % 64 == 0
      async_copy_b128(gB0 + kn, lB0 + lbuf);
      async_copy_b128(gB1 + kn, lB1 + lbuf);
      aO0 = load_frag_a(Arow0 + kn, K);
      aO1 = load_frag_a(Arow1 + kn, K);
#pragma unroll
      for (int j = 0; j < 4; ++j) {
        v16h b = load_frag_b(&Bs[0][(16 * j) * LDB], LDB);
        acc[0][j] = WMMA_F32F16(aE0, b, acc[0][j]);
        acc[1][j] = WMMA_F32F16(aE1, b, acc[1][j]);
      }
      wait_async();
      __syncthreads();
    }
    // ---- phase 1: compute Bs[1] with aO*, stage k0+64 -> Bs[0] ----
    {
      int kn = k0 + 64; if (kn >= K) kn = 0;  // last iter: harmless redundant work
      async_copy_b128(gB0 + kn, lB0);
      async_copy_b128(gB1 + kn, lB1);
      aE0 = load_frag_a(Arow0 + kn, K);
      aE1 = load_frag_a(Arow1 + kn, K);
#pragma unroll
      for (int j = 0; j < 4; ++j) {
        v16h b = load_frag_b(&Bs[1][(16 * j) * LDB], LDB);
        acc[0][j] = WMMA_F32F16(aO0, b, acc[0][j]);
        acc[1][j] = WMMA_F32F16(aO1, b, acc[1][j]);
      }
      wait_async();
      __syncthreads();
    }
  }

#pragma unroll
  for (int s = 0; s < 2; ++s)
#pragma unroll
    for (int j = 0; j < 4; ++j)
#pragma unroll
      for (int r = 0; r < 8; ++r) {
        const int m = m0 + 16 * s + r + 8 * hi;  // C-layout: lanes 0-15 M=r, 16-31 M=r+8
        const int n = n0 + 16 * j + nl;
        const float v = acc[s][j][r] * scale;
        if (out_is_f32) ((float*)out)[(size_t)m * N + n] = v;
        else            ((half_t*)out)[(size_t)m * N + n] = (half_t)v;
      }
}

// ---------------------------------------------------------------------------
// Flash attention, online softmax over the key axis.
// Q,K,V,O: [B, Lx, H*64] f16 (strided views per (b,h)); Q,K pre-scaled.
// Block: 128 threads (4 waves) handles 64 query rows for one (b,h).
// K tiles double-buffered in LDS via async copies; V tile transposed into LDS.
__global__ void flash_attn_wmma(const half_t* __restrict__ Q, const half_t* __restrict__ Km,
                                const half_t* __restrict__ Vm, half_t* __restrict__ O,
                                int Lq, int Lk) {
  constexpr int Cs = 1024;   // row stride in halfs
  constexpr int HD = 64;
  constexpr int LDT = 72;    // padded LDS row stride (halfs): 144B, 16B-aligned

  __shared__ __align__(16) half_t Ks[2][64 * LDT];   // K tile: Ks[s][d], double buffered
  __shared__ __align__(16) half_t Vt[64 * LDT];      // V^T tile: Vt[d][s]
  __shared__ __align__(16) half_t Pl[4 * 16 * LDT];  // per-wave P tile: [16 q rows][64 s]

  const int wave = threadIdx.x >> 5;
  const int lane = threadIdx.x & 31;
  const int hi = lane >> 4, nl = lane & 15;
  const int q0 = blockIdx.x * 64 + wave * 16;
  const int h = blockIdx.y, b = blockIdx.z;

  const half_t* Qb = Q  + (size_t)b * Lq * Cs + h * HD;
  const half_t* Kb = Km + (size_t)b * Lk * Cs + h * HD;
  const half_t* Vb = Vm + (size_t)b * Lk * Cs + h * HD;
  half_t*       Ob = O  + (size_t)b * Lq * Cs + h * HD;

  // this thread's four 16B chunks of a 64x64 K tile (128B per row, 512 chunks)
  const int kc = threadIdx.x;

  // prologue: stage K tile s0=0 into Ks[0]
#pragma unroll
  for (int u = 0; u < 4; ++u) {
    const int c = kc + 128 * u, rw = c >> 3, sg = (c & 7) * 8;
    async_copy_b128(Kb + (size_t)rw * Cs + sg, &Ks[0][rw * LDT + sg]);
  }
  wait_async();
  __syncthreads();

  // Q fragments for the two K-steps over head_dim; kept in VGPRs for the whole loop.
  const v16h aQ0 = load_frag_a(Qb + (size_t)q0 * Cs + 0,  Cs);
  const v16h aQ1 = load_frag_a(Qb + (size_t)q0 * Cs + 32, Cs);

  v8f o[4] = {};
  float rmax[8], rsum[8];
#pragma unroll
  for (int r = 0; r < 8; ++r) { rmax[r] = -INFINITY; rsum[r] = 0.f; }

  half_t* Pw = Pl + wave * 16 * LDT;

  for (int s0 = 0; s0 < Lk; s0 += 64) {
    const int cur = (s0 >> 6) & 1, nxt = cur ^ 1;
    int s0n = s0 + 64; if (s0n >= Lk) s0n = Lk - 64;  // last iter: redundant copy

    // ---- 1) issue async copy of the NEXT K tile into the other buffer ----
#pragma unroll
    for (int u = 0; u < 4; ++u) {
      const int c = kc + 128 * u, rw = c >> 3, sg = (c & 7) * 8;
      async_copy_b128(Kb + (size_t)(s0n + rw) * Cs + sg, &Ks[nxt][rw * LDT + sg]);
    }

    // prefetch next V tile rows into cache (global_prefetch_b8)
    if (s0 + 64 < Lk) {
      const int t = threadIdx.x;
      if (t < 64) __builtin_prefetch(Vb + (size_t)(s0 + 64 + t) * Cs, 0, 0);
    }

    // ---- 2) stage V tile transposed into LDS: Vt[d][s_local] ----
    {
      const int t = threadIdx.x;
      const int sr = t >> 1, dseg = (t & 1) * 32;
      const half_t* vr = Vb + (size_t)(s0 + sr) * Cs + dseg;
#pragma unroll
      for (int d = 0; d < 32; ++d) Vt[(dseg + d) * LDT + sr] = vr[d];
    }

    // ---- 3) sim tile from LDS K: 16 q rows x 64 s cols ----
    v8f sim[4] = {};
#pragma unroll
    for (int j = 0; j < 4; ++j) {
      v16h b0 = load_frag_b(&Ks[cur][(16 * j) * LDT + 0],  LDT);
      sim[j] = WMMA_F32F16(aQ0, b0, sim[j]);
      v16h b1 = load_frag_b(&Ks[cur][(16 * j) * LDT + 32], LDT);
      sim[j] = WMMA_F32F16(aQ1, b1, sim[j]);
    }

    // ---- 4) online softmax update (row-wise over the 64 s cols) ----
#pragma unroll
    for (int r = 0; r < 8; ++r) {
      float mx = fmaxf(fmaxf(sim[0][r], sim[1][r]), fmaxf(sim[2][r], sim[3][r]));
      mx = fmaxf(mx, __shfl_xor(mx, 1));  // reduce within the 16-lane N group
      mx = fmaxf(mx, __shfl_xor(mx, 2));
      mx = fmaxf(mx, __shfl_xor(mx, 4));
      mx = fmaxf(mx, __shfl_xor(mx, 8));
      const float mnew  = fmaxf(rmax[r], mx);
      const float alpha = __expf(rmax[r] - mnew);
      rmax[r] = mnew;
      float ps = 0.f;
#pragma unroll
      for (int j = 0; j < 4; ++j) { sim[j][r] = __expf(sim[j][r] - mnew); ps += sim[j][r]; }
      ps += __shfl_xor(ps, 1);
      ps += __shfl_xor(ps, 2);
      ps += __shfl_xor(ps, 4);
      ps += __shfl_xor(ps, 8);
      rsum[r] = rsum[r] * alpha + ps;
#pragma unroll
      for (int j = 0; j < 4; ++j) o[j][r] *= alpha;
    }

    // ---- 5) stash P (f16) into this wave's LDS region, row-major [m][s] ----
#pragma unroll
    for (int j = 0; j < 4; ++j)
#pragma unroll
      for (int r = 0; r < 8; ++r)
        Pw[(r + 8 * hi) * LDT + 16 * j + nl] = (half_t)sim[j][r];

    // ---- 6) retire async K copy; make Vt/P visible ----
    wait_async();
    __syncthreads();

    // ---- 7) O += P @ V  (contract over 64 s; N = head dim, 4 frags) ----
#pragma unroll
    for (int kk = 0; kk < 2; ++kk) {
      v16h aP = load_frag_a(Pw + kk * 32, LDT);
#pragma unroll
      for (int j = 0; j < 4; ++j) {
        v16h bV = load_frag_b(Vt + (size_t)(16 * j) * LDT + kk * 32, LDT);
        o[j] = WMMA_F32F16(aP, bV, o[j]);
      }
    }

    // ---- 8) protect Vt/P before the next iteration overwrites ----
    __syncthreads();
  }

  // ---- normalize and write O (f16) ----
  float inv[8];
#pragma unroll
  for (int r = 0; r < 8; ++r) inv[r] = 1.f / rsum[r];
#pragma unroll
  for (int j = 0; j < 4; ++j)
#pragma unroll
    for (int r = 0; r < 8; ++r)
      Ob[(size_t)(q0 + r + 8 * hi) * Cs + 16 * j + nl] = (half_t)(o[j][r] * inv[r]);
}

// ---------------------------------------------------------------------------
extern "C" void kernel_launch(void* const* d_in, const int* in_sizes, int n_in,
                              void* d_out, int out_size, void* d_ws, size_t ws_size,
                              hipStream_t stream) {
  (void)in_sizes; (void)n_in; (void)out_size; (void)ws_size;

  constexpr int B = 2, L = 2048, S = 2048, C = 1024, H = 16;
  const size_t nx = (size_t)B * L * C;  // 4,194,304 elems (x0/x1/o0/o1)
  const size_t nw = (size_t)C * C;      // 1,048,576 elems (weights)

  const float* x0  = (const float*)d_in[0];
  const float* x1  = (const float*)d_in[1];
  const float* Wqk = (const float*)d_in[2];
  const float* Wv  = (const float*)d_in[3];
  const float* Wm  = (const float*)d_in[4];

  char* ws = (char*)d_ws;
  half_t* x0h  = (half_t*)ws; ws += nx * sizeof(half_t);
  half_t* x1h  = (half_t*)ws; ws += nx * sizeof(half_t);
  half_t* Wqkh = (half_t*)ws; ws += nw * sizeof(half_t);
  half_t* Wvh  = (half_t*)ws; ws += nw * sizeof(half_t);
  half_t* Wmh  = (half_t*)ws; ws += nw * sizeof(half_t);
  half_t* qk0h = (half_t*)ws; ws += nx * sizeof(half_t);
  half_t* qk1h = (half_t*)ws; ws += nx * sizeof(half_t);
  half_t* v0h  = (half_t*)ws; ws += nx * sizeof(half_t);
  half_t* v1h  = (half_t*)ws; ws += nx * sizeof(half_t);
  half_t* o0h  = (half_t*)ws; ws += nx * sizeof(half_t);
  half_t* o1h  = (half_t*)ws; ws += nx * sizeof(half_t);   // total ~70 MB

  // 1) casts
  cast_f32_to_f16<<<(int)(nx / 2048), 256, 0, stream>>>(x0, x0h, (int)nx);
  cast_f32_to_f16<<<(int)(nx / 2048), 256, 0, stream>>>(x1, x1h, (int)nx);
  cast_f32_to_f16<<<(int)(nw / 2048), 256, 0, stream>>>(Wqk, Wqkh, (int)nw);
  cast_f32_to_f16<<<(int)(nw / 2048), 256, 0, stream>>>(Wv, Wvh, (int)nw);
  cast_f32_to_f16<<<(int)(nw / 2048), 256, 0, stream>>>(Wm, Wmh, (int)nw);

  // 2) projections (scale = 64^-0.25 on both qk streams)
  const float qscale = 0.35355339059327373f;
  const dim3 gg(C / 64, (B * L) / 128);  // (16, 32)
  gemm_nt_wmma<<<gg, 128, 0, stream>>>(x0h, Wqkh, qk0h, B * L, C, C, qscale, 0);
  gemm_nt_wmma<<<gg, 128, 0, stream>>>(x1h, Wqkh, qk1h, B * S, C, C, qscale, 0);
  gemm_nt_wmma<<<gg, 128, 0, stream>>>(x0h, Wvh,  v0h,  B * L, C, C, 1.0f, 0);
  gemm_nt_wmma<<<gg, 128, 0, stream>>>(x1h, Wvh,  v1h,  B * S, C, C, 1.0f, 0);

  // 3) bidirectional attention: softmax over S, then over L (streams swapped)
  const dim3 ga(L / 64, H, B);          // (32, 16, 2)
  flash_attn_wmma<<<ga, 128, 0, stream>>>(qk0h, qk1h, v1h, o0h, L, S);
  flash_attn_wmma<<<ga, 128, 0, stream>>>(qk1h, qk0h, v0h, o1h, S, L);

  // 4) merge projections -> f32 outputs (concatenated: o0 then o1)
  float* out0 = (float*)d_out;
  float* out1 = out0 + nx;
  gemm_nt_wmma<<<gg, 128, 0, stream>>>(o0h, Wmh, (void*)out0, B * L, C, C, 1.0f, 1);
  gemm_nt_wmma<<<gg, 128, 0, stream>>>(o1h, Wmh, (void*)out1, B * S, C, C, 1.0f, 1);
}
